// EdgeAttentionNet_24988119728271
// MI455X (gfx1250) — compile-verified, matrix operand order checked by reading, not run on hardware
//
#include <hip/hip_runtime.h>
#include <hip/hip_bf16.h>

// ---------------------------------------------------------------------------
// Fused EdgeAttentionNet for MI455X (gfx1250, wave32, WMMA f16 16x16x32).
//
// S=1 transformer => softmax(1x1)==1 => MHA collapses to Wo(Wv x + bv) + bo.
//
// One wave owns a 32-edge tile (two stacked 16xK A tiles + two 64-VGPR
// accumulator sets). Each pre-swizzled B fragment (1KB, one contiguous load)
// feeds TWO back-to-back WMMAs -> halves L2 weight traffic and gives ILP to
// hide the weight-load latency; B is additionally double-buffered across the
// column-tile loop. All intermediates live in per-wave LDS f16 tiles.
// ---------------------------------------------------------------------------

typedef __attribute__((ext_vector_type(16))) _Float16 v16h;
typedef __attribute__((ext_vector_type(8)))  float    v8f;

#define E_TOTAL 261120
#define XS 136          // LDS tile stride in halves (32 rows x 136)
#define NW 4            // waves per workgroup (one 32-edge tile per wave)

// ----- param offsets (floats) in jax-pytree sorted-key flatten order -------
#define P_A1_B1   0
#define P_A1_B2   128
#define P_A1_BN_B 256
#define P_A1_BN_G 384
#define P_A1_BN_M 512
#define P_A1_BN_V 640
#define P_A1_W1   768
#define P_A1_W2   17152
#define P_A2_B1   33536
#define P_A2_B2   33664
#define P_A2_BN_B 33792
#define P_A2_BN_G 33920
#define P_A2_BN_M 34048
#define P_A2_BN_V 34176
#define P_A2_W1   34304
#define P_A2_W2   50688
#define P_CLS_B   67072
#define P_CLS_W   67073
#define P_EC_B1   67201
#define P_EC_B2   67329
#define P_EC_W1   67457
#define P_EC_W2   100225
#define P_FC_B    116609
#define P_FC_BN_B 116737
#define P_FC_BN_G 116865
#define P_FC_BN_M 116993
#define P_FC_BN_V 117121
#define P_FC_W    117249
#define P_MS1_B   133633
#define P_MS1_W   133761
#define P_MS2_B   150145
#define P_MS2_W   150273
#define P_MS3_B   183041
#define P_MS3_W   183169
#define P_T_IN_B  232321
#define P_T_IN_W  232705
#define P_T_L1_B  281857
#define P_T_L1_W  282113
#define P_T_L2_B  314881
#define P_T_L2_W  315009
#define P_T_LN1_B 347777
#define P_T_LN1_G 347905
#define P_T_LN2_B 348033
#define P_T_LN2_G 348161
#define P_T_OUT_B 348289
#define P_T_OUT_W 348417

// ----- swizzled weight bases (halves) in workspace -------------------------
#define WB_EC1  0        // K=256
#define WB_EC2  32768    // K=128
#define WB_V    49152    // K=128 (rows 256..383 of t_in_w)
#define WB_OUT  65536    // K=128
#define WB_L1A  81920    // K=128 (t_l1_w rows 0..127)
#define WB_L1B  98304    // K=128 (t_l1_w rows 128..255)
#define WB_L2   114688   // K=256
#define WB_A1G  147456
#define WB_A1S  163840
#define WB_A2G  180224
#define WB_A2S  196608
#define WB_MS1  212992
#define WB_MS2  229376   // K=256
#define WB_MS3  262144   // K=384
#define WB_FC   311296
#define SWZ_TOTAL 327680 // halves
#define BN_BYTE_OFF ((size_t)SWZ_TOTAL * 2)

__device__ __forceinline__ v8f vzero() {
  v8f z = {0.f,0.f,0.f,0.f,0.f,0.f,0.f,0.f};
  return z;
}

// K index for half t (0..15) in a 16-bit WMMA A/B fragment, per ISA layout.
__device__ __host__ __forceinline__ int kmap(int t, int lane) {
  int p = t >> 1, sel = t & 1, hi = (lane >> 4) * 8;
  return (p < 4) ? (hi + 2 * p + sel) : (16 + hi + 2 * (p - 4) + sel);
}

// ---------------------------------------------------------------------------
// Prep kernel: swizzle fp32 weights -> f16 fragment order; fold BN params.
// ---------------------------------------------------------------------------
__global__ void prep_kernel(const float* __restrict__ pr,
                            _Float16* __restrict__ wsw,
                            float* __restrict__ bn) {
  int idx = blockIdx.x * 256 + threadIdx.x;
  if (idx < SWZ_TOTAL) {
    const int WBASE[16] = {0,32768,49152,65536,81920,98304,114688,147456,
                           163840,180224,196608,212992,229376,262144,311296,327680};
    const int SRC[15] = {P_EC_W1,P_EC_W2,P_T_IN_W,P_T_OUT_W,P_T_L1_W,P_T_L1_W,
                         P_T_L2_W,P_A1_W1,P_A1_W2,P_A2_W1,P_A2_W2,P_MS1_W,
                         P_MS2_W,P_MS3_W,P_FC_W};
    const int KS[15]  = {256,128,128,128,128,128,256,128,128,128,128,128,256,384,128};
    const int RO[15]  = {0,0,256,0,0,128,0,0,0,0,0,0,0,0,0};
    int m = 0;
    while (m < 14 && idx >= WBASE[m + 1]) ++m;
    int rel  = idx - WBASE[m];
    int t    = rel & 15;
    int lane = (rel >> 4) & 31;
    int nt   = (rel >> 9) & 7;
    int kt   = rel >> 12;
    int k = kt * 32 + kmap(t, lane);
    int n = nt * 16 + (lane & 15);
    wsw[idx] = (_Float16)pr[SRC[m] + (size_t)(RO[m] + n) * KS[m] + k];
  } else if (idx < SWZ_TOTAL + 384) {
    int j = idx - SWZ_TOTAL;
    int pre = j >> 7, c = j & 127;
    const int GG[3] = {P_A1_BN_G, P_A2_BN_G, P_FC_BN_G};
    const int BB[3] = {P_A1_BN_B, P_A2_BN_B, P_FC_BN_B};
    const int MM[3] = {P_A1_BN_M, P_A2_BN_M, P_FC_BN_M};
    const int VV[3] = {P_A1_BN_V, P_A2_BN_V, P_FC_BN_V};
    float sc = pr[GG[pre] + c] * rsqrtf(pr[VV[pre] + c] + 1e-5f);
    bn[pre * 256 + c]       = sc;
    bn[pre * 256 + 128 + c] = pr[BB[pre] + c] - pr[MM[pre] + c] * sc;
  }
}

// ---------------------------------------------------------------------------
// Dual-M tile GEMM: D[32,128] += A[32, NKT*32] x B.  A from LDS chunks
// (32 rows x 128 cols each), B from pre-swizzled global f16.  Each B
// fragment is consumed by two WMMAs (rows 0-15 and 16-31); B is
// double-buffered across the column-tile loop.
// ---------------------------------------------------------------------------
template <int NKT>
__device__ __forceinline__ void gemm_acc2(const _Float16* const* ch,
                                          const _Float16* __restrict__ w,
                                          int lane, v8f acc0[8], v8f acc1[8]) {
  const int lrow = (lane & 15) * XS;
  const int lhi  = (lane >> 4) * 8;
#pragma unroll
  for (int kt = 0; kt < NKT; ++kt) {
    const _Float16* src = ch[kt >> 2] + (kt & 3) * 32;
    union { v16h h; uint32_t u[8]; } a0, a1;
#pragma unroll
    for (int p = 0; p < 8; ++p) {
      int kb = (p < 4) ? (lhi + 2 * p) : (16 + lhi + 2 * (p - 4));
      a0.u[p] = *(const uint32_t*)(src + lrow + kb);
      a1.u[p] = *(const uint32_t*)(src + 16 * XS + lrow + kb);
    }
    const _Float16* wk = w + (size_t)kt * 4096 + lane * 16;
    union Bf { v16h h; uint4 q[2]; };
    Bf bc, bnx;
    {
      const uint4* bp = (const uint4*)wk;
      bc.q[0] = bp[0];
      bc.q[1] = bp[1];
    }
#pragma unroll
    for (int nt = 0; nt < 8; ++nt) {
      if (nt < 7) {
        const uint4* bp = (const uint4*)(wk + (nt + 1) * 512);
        bnx.q[0] = bp[0];
        bnx.q[1] = bp[1];
      }
      acc0[nt] = __builtin_amdgcn_wmma_f32_16x16x32_f16(
          false, a0.h, false, bc.h, (short)0, acc0[nt], false, false);
      acc1[nt] = __builtin_amdgcn_wmma_f32_16x16x32_f16(
          false, a1.h, false, bc.h, (short)0, acc1[nt], false, false);
      bc = bnx;
    }
  }
}

// Iterate both 16x16 f32 D-fragment layouts (rows 0-15 and 16-31).
template <typename F>
__device__ __forceinline__ void store_tile2(v8f acc0[8], v8f acc1[8], int lane,
                                            F&& f) {
  const int mh = (lane >> 4) << 3;
  const int cl = lane & 15;
#pragma unroll
  for (int nt = 0; nt < 8; ++nt)
#pragma unroll
    for (int i = 0; i < 8; ++i) {
      f(i + mh, nt * 16 + cl, acc0[nt][i]);
      f(i + mh + 16, nt * 16 + cl, acc1[nt][i]);
    }
}

// Per-row LayerNorm: 32 rows, 32 lanes -> fully lane-private, no LDS stats.
__device__ __forceinline__ void layernorm32(_Float16* t, const float* g,
                                            const float* b, int lane) {
  const int base = lane * XS;
  float s = 0.f, s2 = 0.f;
  for (int c = 0; c < 128; ++c) {
    float v = (float)t[base + c];
    s += v; s2 += v * v;
  }
  float mu   = s * (1.f / 128.f);
  float rstd = rsqrtf(s2 * (1.f / 128.f) - mu * mu + 1e-5f);
  for (int c = 0; c < 128; ++c) {
    float v = ((float)t[base + c] - mu) * rstd * g[c] + b[c];
    t[base + c] = (_Float16)v;
  }
}

// ---------------------------------------------------------------------------
// Main fused kernel: one wave = one 32-edge tile, fully resident in LDS.
// ---------------------------------------------------------------------------
__global__ __launch_bounds__(128, 1) void edge_net_kernel(
    const float* __restrict__ pf, const float* __restrict__ pr,
    const int* __restrict__ i1, const int* __restrict__ i2,
    const _Float16* __restrict__ wsw, const float* __restrict__ bn,
    float* __restrict__ out) {
  __shared__ _Float16 s_pf1[NW][32 * XS], s_pf2[NW][32 * XS], s_ef[NW][32 * XS],
      s_t0[NW][32 * XS], s_t1[NW][32 * XS], s_t2[NW][32 * XS],
      s_g[NW][32 * XS], s_att[NW][32 * XS];

  const int lane = threadIdx.x & 31;
  const int w    = threadIdx.x >> 5;
  const int e0   = (blockIdx.x * NW + w) * 32;

  _Float16 *pf1 = s_pf1[w], *pf2 = s_pf2[w], *ef = s_ef[w];
  _Float16 *t0 = s_t0[w], *t1 = s_t1[w], *t2 = s_t2[w];
  _Float16 *gt = s_g[w], *att = s_att[w];

  // ---- gather paired point features (32 rows x 128) ----
  for (int i = lane; i < 4096; i += 32) {
    int r = i >> 7, c = i & 127;
    pf1[r * XS + c] = (_Float16)pf[(size_t)i1[e0 + r] * 128 + c];
    pf2[r * XS + c] = (_Float16)pf[(size_t)i2[e0 + r] * 128 + c];
  }

  v8f acc0[8], acc1[8];
#define ZACC()                                                         \
  _Pragma("unroll") for (int z = 0; z < 8; ++z) {                      \
    acc0[z] = vzero();                                                 \
    acc1[z] = vzero();                                                 \
  }

  // ---- EdgeConv: relu(W1 [pf1|pf2] + b1), relu(W2 h + b2) ----
  {
    const _Float16* ch[2] = {pf1, pf2};
    ZACC();
    gemm_acc2<8>(ch, wsw + WB_EC1, lane, acc0, acc1);
    store_tile2(acc0, acc1, lane, [&](int m, int c, float v) {
      t0[m * XS + c] = (_Float16)fmaxf(v + pr[P_EC_B1 + c], 0.f);
    });
  }
  {
    const _Float16* ch[1] = {t0};
    ZACC();
    gemm_acc2<4>(ch, wsw + WB_EC2, lane, acc0, acc1);
    store_tile2(acc0, acc1, lane, [&](int m, int c, float v) {
      ef[m * XS + c] = (_Float16)fmaxf(v + pr[P_EC_B2 + c], 0.f);
    });
  }

  // ---- Transformer, S=1: MHA == Wo(Wv x + bv) + bo ----
  {
    const _Float16* ch[1] = {ef};
    ZACC();
    gemm_acc2<4>(ch, wsw + WB_V, lane, acc0, acc1);
    store_tile2(acc0, acc1, lane, [&](int m, int c, float v) {
      t0[m * XS + c] = (_Float16)(v + pr[P_T_IN_B + 256 + c]);
    });
  }
  {
    const _Float16* ch[1] = {t0};
    ZACC();
    gemm_acc2<4>(ch, wsw + WB_OUT, lane, acc0, acc1);
    store_tile2(acc0, acc1, lane, [&](int m, int c, float v) {
      t1[m * XS + c] =
          (_Float16)(v + pr[P_T_OUT_B + c] + (float)ef[m * XS + c]);
    });
  }
  layernorm32(t1, pr + P_T_LN1_G, pr + P_T_LN1_B, lane);

  // ---- FFN 128 -> 256 -> 128 with residual + LN ----
  {
    const _Float16* ch[1] = {t1};
    ZACC();
    gemm_acc2<4>(ch, wsw + WB_L1A, lane, acc0, acc1);
    store_tile2(acc0, acc1, lane, [&](int m, int c, float v) {
      t0[m * XS + c] = (_Float16)fmaxf(v + pr[P_T_L1_B + c], 0.f);
    });
  }
  {
    const _Float16* ch[1] = {t1};
    ZACC();
    gemm_acc2<4>(ch, wsw + WB_L1B, lane, acc0, acc1);
    store_tile2(acc0, acc1, lane, [&](int m, int c, float v) {
      t2[m * XS + c] = (_Float16)fmaxf(v + pr[P_T_L1_B + 128 + c], 0.f);
    });
  }
  {
    const _Float16* ch[2] = {t0, t2};
    ZACC();
    gemm_acc2<8>(ch, wsw + WB_L2, lane, acc0, acc1);
    store_tile2(acc0, acc1, lane, [&](int m, int c, float v) {
      gt[m * XS + c] =
          (_Float16)(v + pr[P_T_L2_B + c] + (float)t1[m * XS + c]);
    });
  }
  layernorm32(gt, pr + P_T_LN2_G, pr + P_T_LN2_B, lane);

  // ---- PairedPointAttention: fus = pf1 + pf2; two sigmoid gates; max ----
  for (int i = lane; i < 4096; i += 32) {
    int r = i >> 7, c = i & 127;
    t0[r * XS + c] =
        (_Float16)((float)pf1[r * XS + c] + (float)pf2[r * XS + c]);
  }
  {
    const _Float16* ch[1] = {t0};
    ZACC();
    gemm_acc2<4>(ch, wsw + WB_A1G, lane, acc0, acc1);
    store_tile2(acc0, acc1, lane, [&](int m, int c, float v) {
      t1[m * XS + c] =
          (_Float16)fmaxf((v + pr[P_A1_B1 + c]) * bn[c] + bn[128 + c], 0.f);
    });
  }
  {
    const _Float16* ch[1] = {t1};
    ZACC();
    gemm_acc2<4>(ch, wsw + WB_A1S, lane, acc0, acc1);
    store_tile2(acc0, acc1, lane, [&](int m, int c, float v) {
      float g1 = 1.f / (1.f + __expf(-(v + pr[P_A1_B2 + c])));
      att[m * XS + c] = (_Float16)((float)pf1[m * XS + c] * g1);
    });
  }
  {
    const _Float16* ch[1] = {t0};
    ZACC();
    gemm_acc2<4>(ch, wsw + WB_A2G, lane, acc0, acc1);
    store_tile2(acc0, acc1, lane, [&](int m, int c, float v) {
      t1[m * XS + c] = (_Float16)fmaxf(
          (v + pr[P_A2_B1 + c]) * bn[256 + c] + bn[384 + c], 0.f);
    });
  }
  {
    const _Float16* ch[1] = {t1};
    ZACC();
    gemm_acc2<4>(ch, wsw + WB_A2S, lane, acc0, acc1);
    store_tile2(acc0, acc1, lane, [&](int m, int c, float v) {
      float g2 = 1.f / (1.f + __expf(-(v + pr[P_A2_B2 + c])));
      float a2 = (float)pf2[m * XS + c] * g2;
      att[m * XS + c] = (_Float16)fmaxf((float)att[m * XS + c], a2);
    });
  }

  // ---- edge = ef + global + att ----
  for (int i = lane; i < 4096; i += 32) {
    int r = i >> 7, c = i & 127;
    t0[r * XS + c] = (_Float16)((float)ef[r * XS + c] +
                                (float)gt[r * XS + c] +
                                (float)att[r * XS + c]);
  }

  // ---- MultiScaleFusion ----
  {
    const _Float16* ch[1] = {t0};
    ZACC();
    gemm_acc2<4>(ch, wsw + WB_MS1, lane, acc0, acc1);
    store_tile2(acc0, acc1, lane, [&](int m, int c, float v) {
      t1[m * XS + c] = (_Float16)(v + pr[P_MS1_B + c]);
    });
  }
  {
    const _Float16* ch[2] = {t0, t1};
    ZACC();
    gemm_acc2<8>(ch, wsw + WB_MS2, lane, acc0, acc1);
    store_tile2(acc0, acc1, lane, [&](int m, int c, float v) {
      t2[m * XS + c] = (_Float16)(v + pr[P_MS2_B + c]);
    });
  }
  {
    const _Float16* ch[3] = {t0, t1, t2};
    ZACC();
    gemm_acc2<12>(ch, wsw + WB_MS3, lane, acc0, acc1);
    store_tile2(acc0, acc1, lane, [&](int m, int c, float v) {
      pf1[m * XS + c] = (_Float16)(v + pr[P_MS3_B + c]);  // reuse pf1 tile
    });
  }

  // ---- shared_fc (Linear + BN + ReLU) ----
  {
    const _Float16* ch[1] = {pf1};
    ZACC();
    gemm_acc2<4>(ch, wsw + WB_FC, lane, acc0, acc1);
    store_tile2(acc0, acc1, lane, [&](int m, int c, float v) {
      t0[m * XS + c] = (_Float16)fmaxf(
          (v + pr[P_FC_B + c]) * bn[512 + c] + bn[640 + c], 0.f);
    });
  }

  // ---- classifier + sigmoid (one lane per row, 32 rows) ----
  {
    float s = pr[P_CLS_B];
    for (int c = 0; c < 128; ++c)
      s += (float)t0[lane * XS + c] * pr[P_CLS_W + c];
    out[e0 + lane] = 1.f / (1.f + __expf(-s));
  }
#undef ZACC
}

// ---------------------------------------------------------------------------
extern "C" void kernel_launch(void* const* d_in, const int* in_sizes, int n_in,
                              void* d_out, int out_size, void* d_ws,
                              size_t ws_size, hipStream_t stream) {
  (void)in_sizes; (void)n_in; (void)out_size; (void)ws_size;
  const float* pf = (const float*)d_in[0];
  const float* pr = (const float*)d_in[1];
  const int*   i1 = (const int*)d_in[2];
  const int*   i2 = (const int*)d_in[3];
  _Float16* wsw = (_Float16*)d_ws;
  float*    bn  = (float*)((char*)d_ws + BN_BYTE_OFF);
  float*    out = (float*)d_out;

  int prep_threads = SWZ_TOTAL + 384;
  prep_kernel<<<(prep_threads + 255) / 256, 256, 0, stream>>>(pr, wsw, bn);

  int grid = E_TOTAL / (32 * NW);  // 2040, exact
  edge_net_kernel<<<grid, 128, 0, stream>>>(pf, pr, i1, i2, wsw, bn, out);
}
